// Block_31722628448422
// MI455X (gfx1250) — matrix operand order repaired and checked by reading, not compiled, not run on hardware
//
#include <hip/hip_runtime.h>
#include <hip/hip_bf16.h>
#include <math.h>
#include <stdint.h>

// ---------------------------------------------------------------------------
// Types for WMMA fragments (gfx1250, wave32)
// ---------------------------------------------------------------------------
typedef __attribute__((ext_vector_type(16))) _Float16 v16h;
typedef __attribute__((ext_vector_type(8)))  _Float16 v8h;
typedef __attribute__((ext_vector_type(8)))  float    v8f;
typedef __attribute__((ext_vector_type(4)))  uint32_t u32x4;
typedef __attribute__((ext_vector_type(8)))  uint32_t u32x8;

__device__ __forceinline__ v8f wmma_f16(v16h a, v16h b, v8f c) {
    // D = A(16x32 f16) * B(32x16 f16) + C(16x16 f32)
    return __builtin_amdgcn_wmma_f32_16x16x32_f16(
        /*neg_a=*/false, a, /*neg_b=*/false, b,
        /*c_mod=*/(short)0, c, /*reuse_a=*/false, /*reuse_b=*/false);
}

// A-fragment (16x32, f16, row-major source, row pointer given).
// Layout used here (contiguous k, B loaded the same way): lane l -> row m = l&15, group g = l>>4;
// elements a[0..7]  = k = g*16 + (0..7)
// elements a[8..15] = k = g*16 + 8 + (0..7)
__device__ __forceinline__ v16h load_a_frag(const _Float16* __restrict__ row,
                                            int k0, int g) {
    const v8h lo = *(const v8h*)(row + k0 + g * 16);
    const v8h hi = *(const v8h*)(row + k0 + g * 16 + 8);
    v16h r;
#pragma unroll
    for (int i = 0; i < 8; ++i) { r[i] = lo[i]; r[i + 8] = hi[i]; }
    return r;
}

// ---------------------------------------------------------------------------
// Async global -> LDS staging (gfx1250, ASYNCcnt-tracked)
// ---------------------------------------------------------------------------
__device__ __forceinline__ void async_ld_b128(void* lds_dst, const void* gsrc) {
    // Low 32 bits of a generic LDS pointer == LDS byte offset on AMDGPU.
    uint32_t dst = (uint32_t)(uintptr_t)lds_dst;
    asm volatile("global_load_async_to_lds_b128 %0, %1, off"
                 :: "v"(dst), "v"(gsrc) : "memory");
}

template <int N>
__device__ __forceinline__ void wait_async() {
    asm volatile("s_wait_asynccnt %0" :: "n"(N) : "memory");
}

// ---------------------------------------------------------------------------
// Tensor Data Mover: 2D tile load global -> LDS (TENSORcnt-tracked).
// D# per CDNA5 ISA ch.8: group0 {count, lds_addr, global_addr, type=2},
// group1 {data_size=2B, tensor_dim0/1, tile_dim0/1, tensor_dim0_stride}.
// All fields uniform; issue from one wave, wait via s_wait_tensorcnt.
// ---------------------------------------------------------------------------
__device__ __forceinline__ void tdm_load_2d(uint32_t lds_addr, const void* gaddr,
                                            uint32_t rowlen_elems,  // tensor_dim0
                                            uint32_t tile_w_elems,  // tile_dim0
                                            uint32_t tile_h,        // tile_dim1
                                            uint32_t stride_elems)  // dim0 stride
{
    const uint64_t ga = (uint64_t)(uintptr_t)gaddr;
    u32x4 g0;
    g0[0] = 1u;                                   // count=1, user mode
    g0[1] = lds_addr;                             // LDS byte address
    g0[2] = (uint32_t)ga;                         // global_addr[31:0]
    g0[3] = (uint32_t)((ga >> 32) & 0x1FFFFFFu)   // global_addr[56:32]
          | (2u << 30);                           // type=2 ("image")
    u32x8 g1;
    g1[0] = 1u << 16;                             // data_size=1 (2 bytes)
    g1[1] = (rowlen_elems & 0xFFFFu) << 16;       // tensor_dim0[15:0]
    g1[2] = (rowlen_elems >> 16)                  // tensor_dim0[31:16]
          | ((tile_h & 0xFFFFu) << 16);           // tensor_dim1[15:0]
    g1[3] = (tile_w_elems & 0xFFFFu) << 16;       // tile_dim0
    g1[4] = tile_h & 0xFFFFu;                     // tile_dim1 (tile_dim2=0)
    g1[5] = stride_elems;                         // tensor_dim0_stride[31:0]
    g1[6] = 0;                                    // stride hi / dim1_stride lo
    g1[7] = 0;
    asm volatile("tensor_load_to_lds %0, %1" :: "s"(g0), "s"(g1) : "memory");
}

// ---------------------------------------------------------------------------
// Weight conversion kernels (fp32 -> f16, transposed to [N,K])
// ---------------------------------------------------------------------------
__global__ void cvt_transpose_kernel(const float* __restrict__ W,
                                     _Float16* __restrict__ Wt,
                                     int K, int N) {
    size_t idx = (size_t)blockIdx.x * blockDim.x + threadIdx.x;
    size_t total = (size_t)K * N;
    if (idx >= total) return;
    size_t n = idx / K, k = idx % K;
    Wt[idx] = (_Float16)W[k * (size_t)N + n];
}

// W: [H,E,DH] fp32 -> Wt: [H*DH, E] f16  (Wt[(h*DH+d)*E+e] = W[h,e,d])
__global__ void cvt_headproj_kernel(const float* __restrict__ W,
                                    _Float16* __restrict__ Wt,
                                    int E, int H, int DH) {
    size_t idx = (size_t)blockIdx.x * blockDim.x + threadIdx.x;
    size_t total = (size_t)H * E * DH;
    if (idx >= total) return;
    size_t n = idx / E, e = idx % E;
    size_t h = n / DH, d = n % DH;
    Wt[idx] = (_Float16)W[(h * E + e) * (size_t)DH + d];
}

// ---------------------------------------------------------------------------
// LayerNorm: one 256-thread block per row; writes f16
// ---------------------------------------------------------------------------
__global__ void __launch_bounds__(256)
ln_kernel(const float* __restrict__ X, const float* __restrict__ gam,
          const float* __restrict__ bet, _Float16* __restrict__ Y, int E) {
    __shared__ float s1[256];
    __shared__ float s2[256];
    const int row = blockIdx.x;
    const int tid = threadIdx.x;
    const float* x = X + (size_t)row * E;
    float a = 0.f, b = 0.f;
    for (int i = tid; i < E; i += 256) { float v = x[i]; a += v; b += v * v; }
    s1[tid] = a; s2[tid] = b;
    __syncthreads();
    for (int st = 128; st > 0; st >>= 1) {
        if (tid < st) { s1[tid] += s1[tid + st]; s2[tid] += s2[tid + st]; }
        __syncthreads();
    }
    const float mu  = s1[0] / (float)E;
    const float var = s2[0] / (float)E - mu * mu;
    const float rs  = rsqrtf(var + 1e-5f);
    _Float16* y = Y + (size_t)row * E;
    for (int i = tid; i < E; i += 256)
        y[i] = (_Float16)((x[i] - mu) * rs * gam[i] + bet[i]);
}

// ---------------------------------------------------------------------------
// WMMA GEMM, f16 output.  C = alpha * A[M,K] @ Wt[N,K]^T
// mode 0: Out[m*N+n]; mode 1: V^T layout Out[((b*H+h)*DH+d)*S + s].
// 256 threads = 8 waves; wave tile 32x64; block tile 256x64.
// B staged by TDM: 64x128-half chunks, double buffered (32 KB LDS),
// 4 K-steps (32 WMMAs/wave) per barrier pair.
// ---------------------------------------------------------------------------
__global__ void __launch_bounds__(256)
gemm_f16_kernel(const _Float16* __restrict__ A, const _Float16* __restrict__ Wt,
                _Float16* __restrict__ Out, int M, int N, int K,
                float alpha, int mode, int S, int H, int DH) {
    __shared__ __align__(16) _Float16 Bt[2][64 * 128];

    const int tid  = threadIdx.x;
    const int lane = tid & 31;
    const int wave = tid >> 5;
    const int g    = lane >> 4;
    const int ln   = lane & 15;
    const int m0   = blockIdx.y * 256 + wave * 32;
    const int n0   = blockIdx.x * 64;

    v8f acc[2][4];
    v8f z = {};
#pragma unroll
    for (int s = 0; s < 2; ++s)
#pragma unroll
        for (int t = 0; t < 4; ++t) acc[s][t] = z;

    const _Float16* Arow0 = A + (size_t)(m0 + ln) * K;
    const _Float16* Arow1 = A + (size_t)(m0 + 16 + ln) * K;
    const _Float16* Wtile = Wt + (size_t)n0 * K;   // row n0, k=0

    auto compute_chunk = [&](int c) {
        const _Float16* bb = &Bt[c & 1][0];
#pragma unroll
        for (int j = 0; j < 4; ++j) {
            const int kc = c * 128 + j * 32;
            const v16h af0 = load_a_frag(Arow0, kc, g);
            const v16h af1 = load_a_frag(Arow1, kc, g);
#pragma unroll
            for (int t = 0; t < 4; ++t) {
                const v16h bf =
                    *(const v16h*)(bb + (t * 16 + ln) * 128 + j * 32 + g * 16);
                acc[0][t] = wmma_f16(af0, bf, acc[0][t]);
                acc[1][t] = wmma_f16(af1, bf, acc[1][t]);
            }
        }
    };

    const int nc = K / 128;
    const bool issuer = (wave == 0);
    if (issuer)
        tdm_load_2d((uint32_t)(uintptr_t)&Bt[0][0], Wtile, K, 128, 64, K);
    for (int c = 0; c < nc - 1; ++c) {
        if (issuer) {
            tdm_load_2d((uint32_t)(uintptr_t)&Bt[(c + 1) & 1][0],
                        Wtile + (size_t)(c + 1) * 128, K, 128, 64, K);
            __builtin_amdgcn_s_wait_tensorcnt((short)1);
        }
        __syncthreads();
        compute_chunk(c);
        __syncthreads();
    }
    if (issuer) __builtin_amdgcn_s_wait_tensorcnt((short)0);
    __syncthreads();
    compute_chunk(nc - 1);

#pragma unroll
    for (int s = 0; s < 2; ++s) {
#pragma unroll
        for (int t = 0; t < 4; ++t) {
#pragma unroll
            for (int v = 0; v < 8; ++v) {
                const int m = m0 + s * 16 + v + 8 * g;
                const int n = n0 + t * 16 + ln;
                const _Float16 val = (_Float16)(alpha * acc[s][t][v]);
                if (mode == 0) {
                    Out[(size_t)m * N + n] = val;
                } else {
                    const int b = m / S, ss = m % S;
                    const int h = n / DH, d = n % DH;
                    Out[(((size_t)b * H + h) * DH + d) * (size_t)S + ss] = val;
                }
            }
        }
    }
}

// ---------------------------------------------------------------------------
// WMMA GEMM, fp32 output with bias + residual:
//   Out[m,n] = A@Wt + bias[n] + Res[m,n]
// Same TDM-staged structure as gemm_f16_kernel.
// ---------------------------------------------------------------------------
__global__ void __launch_bounds__(256)
gemm_bias_res_kernel(const _Float16* __restrict__ A,
                     const _Float16* __restrict__ Wt,
                     const float* __restrict__ bias,
                     const float* __restrict__ Res,
                     float* __restrict__ Out, int M, int N, int K) {
    __shared__ __align__(16) _Float16 Bt[2][64 * 128];

    const int tid  = threadIdx.x;
    const int lane = tid & 31;
    const int wave = tid >> 5;
    const int g    = lane >> 4;
    const int ln   = lane & 15;
    const int m0   = blockIdx.y * 256 + wave * 32;
    const int n0   = blockIdx.x * 64;

    v8f acc[2][4];
    v8f z = {};
#pragma unroll
    for (int s = 0; s < 2; ++s)
#pragma unroll
        for (int t = 0; t < 4; ++t) acc[s][t] = z;

    const _Float16* Arow0 = A + (size_t)(m0 + ln) * K;
    const _Float16* Arow1 = A + (size_t)(m0 + 16 + ln) * K;
    const _Float16* Wtile = Wt + (size_t)n0 * K;

    auto compute_chunk = [&](int c) {
        const _Float16* bb = &Bt[c & 1][0];
#pragma unroll
        for (int j = 0; j < 4; ++j) {
            const int kc = c * 128 + j * 32;
            const v16h af0 = load_a_frag(Arow0, kc, g);
            const v16h af1 = load_a_frag(Arow1, kc, g);
#pragma unroll
            for (int t = 0; t < 4; ++t) {
                const v16h bf =
                    *(const v16h*)(bb + (t * 16 + ln) * 128 + j * 32 + g * 16);
                acc[0][t] = wmma_f16(af0, bf, acc[0][t]);
                acc[1][t] = wmma_f16(af1, bf, acc[1][t]);
            }
        }
    };

    const int nc = K / 128;
    const bool issuer = (wave == 0);
    if (issuer)
        tdm_load_2d((uint32_t)(uintptr_t)&Bt[0][0], Wtile, K, 128, 64, K);
    for (int c = 0; c < nc - 1; ++c) {
        if (issuer) {
            tdm_load_2d((uint32_t)(uintptr_t)&Bt[(c + 1) & 1][0],
                        Wtile + (size_t)(c + 1) * 128, K, 128, 64, K);
            __builtin_amdgcn_s_wait_tensorcnt((short)1);
        }
        __syncthreads();
        compute_chunk(c);
        __syncthreads();
    }
    if (issuer) __builtin_amdgcn_s_wait_tensorcnt((short)0);
    __syncthreads();
    compute_chunk(nc - 1);

#pragma unroll
    for (int s = 0; s < 2; ++s) {
#pragma unroll
        for (int t = 0; t < 4; ++t) {
#pragma unroll
            for (int v = 0; v < 8; ++v) {
                const int m = m0 + s * 16 + v + 8 * g;
                const int n = n0 + t * 16 + ln;
                Out[(size_t)m * N + n] =
                    acc[s][t][v] + bias[n] + Res[(size_t)m * N + n];
            }
        }
    }
}

// ---------------------------------------------------------------------------
// Fused SwiGLU GEMM: u = silu(A@W1t + b1) * (A@W2t + b2), f16 out.
// Stages BOTH weight tiles via async global->LDS b128 (ASYNCcnt path),
// double buffered, branch-free steady-state loop.
// 256 threads = 8 waves; wave tile 16x64; block tile 128x64.
// ---------------------------------------------------------------------------
__global__ void __launch_bounds__(256)
gemm_swiglu_kernel(const _Float16* __restrict__ A,
                   const _Float16* __restrict__ W1t,
                   const _Float16* __restrict__ W2t,
                   const float* __restrict__ b1,
                   const float* __restrict__ b2,
                   _Float16* __restrict__ U, int M, int N, int K) {
    __shared__ __align__(16) _Float16 B1[2][64 * 32];
    __shared__ __align__(16) _Float16 B2[2][64 * 32];

    const int tid  = threadIdx.x;
    const int lane = tid & 31;
    const int wave = tid >> 5;
    const int g    = lane >> 4;
    const int ln   = lane & 15;
    const int m0   = blockIdx.y * 128 + wave * 16;
    const int n0   = blockIdx.x * 64;

    const int ldn = tid >> 2;
    const int ldc = (tid & 3) * 8;
    const _Float16* W1row = W1t + (size_t)(n0 + ldn) * K + ldc;
    const _Float16* W2row = W2t + (size_t)(n0 + ldn) * K + ldc;

    v8f acc1[4], acc2[4];
    v8f z = {};
#pragma unroll
    for (int t = 0; t < 4; ++t) { acc1[t] = z; acc2[t] = z; }

    const _Float16* Arow = A + (size_t)(m0 + ln) * K;

    auto compute = [&](int i) {
        const int kc = i * 32;
        const v16h af = load_a_frag(Arow, kc, g);
        const _Float16* bb1 = &B1[i & 1][0];
        const _Float16* bb2 = &B2[i & 1][0];
#pragma unroll
        for (int t = 0; t < 4; ++t) {
            const v16h bf1 = *(const v16h*)(bb1 + (t * 16 + ln) * 32 + g * 16);
            acc1[t] = wmma_f16(af, bf1, acc1[t]);
            const v16h bf2 = *(const v16h*)(bb2 + (t * 16 + ln) * 32 + g * 16);
            acc2[t] = wmma_f16(af, bf2, acc2[t]);
        }
    };

    const int nk = K / 32;
    async_ld_b128(&B1[0][ldn * 32 + ldc], W1row);
    async_ld_b128(&B2[0][ldn * 32 + ldc], W2row);
    for (int i = 0; i < nk - 1; ++i) {
        async_ld_b128(&B1[(i + 1) & 1][ldn * 32 + ldc], W1row + (i + 1) * 32);
        async_ld_b128(&B2[(i + 1) & 1][ldn * 32 + ldc], W2row + (i + 1) * 32);
        wait_async<2>();
        __syncthreads();
        compute(i);
        __syncthreads();
    }
    wait_async<0>();
    __syncthreads();
    compute(nk - 1);

#pragma unroll
    for (int t = 0; t < 4; ++t) {
#pragma unroll
        for (int v = 0; v < 8; ++v) {
            const int m = m0 + v + 8 * g;
            const int n = n0 + t * 16 + ln;
            const float a  = acc1[t][v] + b1[n];
            const float gg = acc2[t][v] + b2[n];
            const float sig = 1.0f / (1.0f + __expf(-a));
            U[(size_t)m * N + n] = (_Float16)(a * sig * gg);
        }
    }
}

// ---------------------------------------------------------------------------
// Flash attention (causal, online softmax). One wave per 16-row query tile.
// Q,K stored [B*S, E] f16 (Q pre-scaled by DH^-0.5); V stored transposed
// [B,H,DH,S] f16. Output ctx [B*S, E] f16.
// ---------------------------------------------------------------------------
__global__ void __launch_bounds__(32)
attn_kernel(const _Float16* __restrict__ Q, const _Float16* __restrict__ Kk,
            const _Float16* __restrict__ Vt, _Float16* __restrict__ Ctx,
            int B, int H, int S, int E, int DH) {
    __shared__ __align__(16) _Float16 Pl[16 * 32];

    const int lane = threadIdx.x & 31;
    const int g    = lane >> 4;
    const int ln   = lane & 15;
    const int q0   = blockIdx.x * 16;
    const int h    = blockIdx.y;
    const int b    = blockIdx.z;

    const _Float16* Qb = Q  + (size_t)b * S * E + (size_t)h * DH;
    const _Float16* Kb = Kk + (size_t)b * S * E + (size_t)h * DH;
    const _Float16* Vb = Vt + ((size_t)b * H + h) * DH * (size_t)S;

    const _Float16* qrow = Qb + (size_t)(q0 + ln) * E;
    const v16h qf0 = load_a_frag(qrow, 0, g);
    const v16h qf1 = load_a_frag(qrow, 32, g);

    v8f acc[4];
    v8f z = {};
#pragma unroll
    for (int t = 0; t < 4; ++t) acc[t] = z;
    float mrow[8], lrow[8];
#pragma unroll
    for (int v = 0; v < 8; ++v) { mrow[v] = -__builtin_inff(); lrow[v] = 0.f; }

    const int nk = q0 + 16;                 // causal: keys [0, q0+16)
    for (int sk0 = 0; sk0 < nk; sk0 += 32) {
        // ---- scores: two 16x16 tiles over this 32-key block ----
        v8f sc[2];
#pragma unroll
        for (int t = 0; t < 2; ++t) {
            const _Float16* krow = Kb + (size_t)(sk0 + t * 16 + ln) * E;
            const v16h kf0 = *(const v16h*)(krow + g * 16);        // d 0..31
            const v16h kf1 = *(const v16h*)(krow + 32 + g * 16);   // d 32..63
            v8f s = z;
            s = wmma_f16(qf0, kf0, s);
            s = wmma_f16(qf1, kf1, s);
            sc[t] = s;
        }

        // ---- mask + online softmax stats, stage P to LDS ----
#pragma unroll
        for (int v = 0; v < 8; ++v) {
            const int sq = q0 + v + 8 * g;
            float s0 = (sk0 + ln      <= sq) ? sc[0][v] : -__builtin_inff();
            float s1 = (sk0 + 16 + ln <= sq) ? sc[1][v] : -__builtin_inff();
            float mx = fmaxf(s0, s1);
            mx = fmaxf(mx, __shfl_xor(mx, 1, 32));
            mx = fmaxf(mx, __shfl_xor(mx, 2, 32));
            mx = fmaxf(mx, __shfl_xor(mx, 4, 32));
            mx = fmaxf(mx, __shfl_xor(mx, 8, 32));
            const float mnew = fmaxf(mrow[v], mx);
            const float corr = __expf(mrow[v] - mnew);
            const float p0 = __expf(s0 - mnew);
            const float p1 = __expf(s1 - mnew);
            float ps = p0 + p1;
            ps += __shfl_xor(ps, 1, 32);
            ps += __shfl_xor(ps, 2, 32);
            ps += __shfl_xor(ps, 4, 32);
            ps += __shfl_xor(ps, 8, 32);
            lrow[v] = lrow[v] * corr + ps;
            mrow[v] = mnew;
#pragma unroll
            for (int t = 0; t < 4; ++t) acc[t][v] *= corr;
            Pl[(v + 8 * g) * 32 + ln]      = (_Float16)p0;
            Pl[(v + 8 * g) * 32 + 16 + ln] = (_Float16)p1;
        }
        __syncthreads();

        // ---- P (16x32 from LDS) @ V (32x64) ----
        const v16h pf = load_a_frag(Pl + (size_t)ln * 32, 0, g);
#pragma unroll
        for (int t = 0; t < 4; ++t) {
            const _Float16* vrow = Vb + (size_t)(t * 16 + ln) * S + sk0 + g * 16;
            const v16h vf = *(const v16h*)vrow;
            acc[t] = wmma_f16(pf, vf, acc[t]);
        }
        __syncthreads();
    }

    // ---- normalize and store ----
#pragma unroll
    for (int t = 0; t < 4; ++t) {
#pragma unroll
        for (int v = 0; v < 8; ++v) {
            const int m = q0 + v + 8 * g;
            const float r = acc[t][v] / lrow[v];
            Ctx[((size_t)b * S + m) * E + (size_t)h * DH + t * 16 + ln] =
                (_Float16)r;
        }
    }
}

// ---------------------------------------------------------------------------
// Host-side launcher
// ---------------------------------------------------------------------------
extern "C" void kernel_launch(void* const* d_in, const int* in_sizes, int n_in,
                              void* d_out, int out_size, void* d_ws,
                              size_t ws_size, hipStream_t stream) {
    (void)in_sizes; (void)n_in; (void)out_size; (void)ws_size;

    constexpr int Bn = 4, Sn = 2048, En = 1024, Hn = 16, DHn = 64, DFn = 4096;
    constexpr int Mn = Bn * Sn;   // 8192 rows

    const float* x   = (const float*)d_in[0];
    const float* wq  = (const float*)d_in[1];
    const float* wk  = (const float*)d_in[2];
    const float* wv  = (const float*)d_in[3];
    const float* wo  = (const float*)d_in[4];
    const float* bo  = (const float*)d_in[5];
    const float* w1  = (const float*)d_in[6];
    const float* b1  = (const float*)d_in[7];
    const float* w2  = (const float*)d_in[8];
    const float* b2  = (const float*)d_in[9];
    const float* w3  = (const float*)d_in[10];
    const float* b3  = (const float*)d_in[11];
    const float* g1  = (const float*)d_in[12];
    const float* be1 = (const float*)d_in[13];
    const float* g2  = (const float*)d_in[14];
    const float* be2 = (const float*)d_in[15];
    float* out = (float*)d_out;

    // workspace carve-up (256B aligned slices)
    char* ws = (char*)d_ws;
    size_t off = 0;
    auto carve = [&](size_t bytes) -> void* {
        void* p = ws + off;
        off = (off + bytes + 255) & ~(size_t)255;
        return p;
    };
    _Float16* wqT = (_Float16*)carve((size_t)En * En * 2);
    _Float16* wkT = (_Float16*)carve((size_t)En * En * 2);
    _Float16* wvT = (_Float16*)carve((size_t)En * En * 2);
    _Float16* woT = (_Float16*)carve((size_t)En * En * 2);
    _Float16* w1T = (_Float16*)carve((size_t)En * DFn * 2);
    _Float16* w2T = (_Float16*)carve((size_t)En * DFn * 2);
    _Float16* w3T = (_Float16*)carve((size_t)DFn * En * 2);
    _Float16* h16 = (_Float16*)carve((size_t)Mn * En * 2);
    _Float16* q16 = (_Float16*)carve((size_t)Mn * En * 2);
    _Float16* k16 = (_Float16*)carve((size_t)Mn * En * 2);
    _Float16* vT  = (_Float16*)carve((size_t)Mn * En * 2);
    _Float16* ctx = (_Float16*)carve((size_t)Mn * En * 2);
    float*    r1  = (float*)   carve((size_t)Mn * En * 4);
    _Float16* h2  = (_Float16*)carve((size_t)Mn * En * 2);
    _Float16* u16 = (_Float16*)carve((size_t)Mn * DFn * 2);

    // ---- 1. weight conversion ----
    {
        const int thr = 256;
        size_t nEE = (size_t)En * En;
        size_t nEF = (size_t)En * DFn;
        cvt_headproj_kernel<<<dim3((nEE + thr - 1) / thr), thr, 0, stream>>>(wq, wqT, En, Hn, DHn);
        cvt_headproj_kernel<<<dim3((nEE + thr - 1) / thr), thr, 0, stream>>>(wk, wkT, En, Hn, DHn);
        cvt_headproj_kernel<<<dim3((nEE + thr - 1) / thr), thr, 0, stream>>>(wv, wvT, En, Hn, DHn);
        cvt_transpose_kernel<<<dim3((nEE + thr - 1) / thr), thr, 0, stream>>>(wo, woT, En, En);
        cvt_transpose_kernel<<<dim3((nEF + thr - 1) / thr), thr, 0, stream>>>(w1, w1T, En, DFn);
        cvt_transpose_kernel<<<dim3((nEF + thr - 1) / thr), thr, 0, stream>>>(w2, w2T, En, DFn);
        cvt_transpose_kernel<<<dim3((nEF + thr - 1) / thr), thr, 0, stream>>>(w3, w3T, DFn, En);
    }

    // ---- 2. LN1 ----
    ln_kernel<<<dim3(Mn), 256, 0, stream>>>(x, g1, be1, h16, En);

    // ---- 3. QKV projections ----
    const float qscale = 0.125f;  // DH^-0.5 = 1/8
    dim3 gEE(En / 64, Mn / 256);
    gemm_f16_kernel<<<gEE, 256, 0, stream>>>(h16, wqT, q16, Mn, En, En, qscale, 0, Sn, Hn, DHn);
    gemm_f16_kernel<<<gEE, 256, 0, stream>>>(h16, wkT, k16, Mn, En, En, 1.0f, 0, Sn, Hn, DHn);
    gemm_f16_kernel<<<gEE, 256, 0, stream>>>(h16, wvT, vT,  Mn, En, En, 1.0f, 1, Sn, Hn, DHn);

    // ---- 4. causal flash attention ----
    attn_kernel<<<dim3(Sn / 16, Hn, Bn), 32, 0, stream>>>(q16, k16, vT, ctx,
                                                          Bn, Hn, Sn, En, DHn);

    // ---- 5. output projection + bias + residual -> r1 ----
    gemm_bias_res_kernel<<<gEE, 256, 0, stream>>>(ctx, woT, bo, x, r1, Mn, En, En);

    // ---- 6. LN2 ----
    ln_kernel<<<dim3(Mn), 256, 0, stream>>>(r1, g2, be2, h2, En);

    // ---- 7. fused SwiGLU ----
    dim3 gEF(DFn / 64, Mn / 128);
    gemm_swiglu_kernel<<<gEF, 256, 0, stream>>>(h2, w1T, w2T, b1, b2, u16, Mn, DFn, En);

    // ---- 8. down projection + bias + residual -> out ----
    gemm_bias_res_kernel<<<gEE, 256, 0, stream>>>(u16, w3T, b3, r1, out, Mn, En, DFn);
}
